// PRSNet_87746181857976
// MI455X (gfx1250) — compile-verified, matrix operand order checked by reading, not run on hardware
//
#include <hip/hip_runtime.h>
#include <hip/hip_bf16.h>
#include <cstdint>

// ---------------- problem constants ----------------
#define BSZ     8
#define NG      19836
#define DIN     11
#define DH      64
#define NN      (BSZ * NG)        // 158688 nodes
#define EE      (NG * 16 * BSZ)   // 2539008 edges
#define NT      (NN / 16)         // 9918 16-row tiles (exact)

// output layout: preds[8], w[NN], sae_loss[1], dec[NN*64]
#define OUT_W_OFF    8
#define OUT_LOSS_OFF (8 + NN)
#define OUT_DEC_OFF  (8 + NN + 1)

typedef _Float16 v16h __attribute__((ext_vector_type(16)));
typedef _Float16 h2   __attribute__((ext_vector_type(2)));
typedef float    v8f  __attribute__((ext_vector_type(8)));

__device__ __forceinline__ float gelu_exact(float x) {
    return 0.5f * x * (1.0f + erff(x * 0.70710678118654752f));
}

// K-index for fragment element group v (0..7), halfsel = 0 (lanes 0-15) or 8 (lanes 16-31)
__device__ __forceinline__ int frag_k(int v, int halfsel) {
    return ((v < 4) ? (2 * v) : (16 + 2 * (v - 4))) + halfsel;
}

__device__ __forceinline__ v8f wmma16(v16h a, v16h b, v8f c) {
    return __builtin_amdgcn_wmma_f32_16x16x32_f16(false, a, false, b, (short)0, c, false, false);
}

// Load B fragment (32x16 slice of 64x64 weight, W[j*64+k]) from LDS, cols [colbase,colbase+16)
__device__ __forceinline__ v16h load_b_lds(const _Float16* W, int lane, int colbase, int chunk) {
    int l = lane & 15, hb = (lane >> 4) * 8;
    const _Float16* p = W + (colbase + l) * DH;
    v16h b;
#pragma unroll
    for (int v = 0; v < 8; ++v) {
        int k = frag_k(v, hb) + 32 * chunk;
        h2 t = *(const h2*)(p + k);
        b[2 * v] = t[0]; b[2 * v + 1] = t[1];
    }
    return b;
}

// Load A fragment (16x32 slice) from an f32 row-major [16][64] tile in global memory
__device__ __forceinline__ v16h load_a_gf32(const float* rowbase, int lane, int chunk) {
    int l = lane & 15, hb = (lane >> 4) * 8;
    const float* p = rowbase + l * DH;
    v16h a;
#pragma unroll
    for (int v = 0; v < 8; ++v) {
        int k = frag_k(v, hb) + 32 * chunk;
        float2 t = *(const float2*)(p + k);
        a[2 * v] = (_Float16)t.x; a[2 * v + 1] = (_Float16)t.y;
    }
    return a;
}

// Load A fragment from an f16 row-major [16][64] tile in LDS
__device__ __forceinline__ v16h load_a_lds(const _Float16* z, int lane, int chunk) {
    int l = lane & 15, hb = (lane >> 4) * 8;
    const _Float16* p = z + l * DH;
    v16h a;
#pragma unroll
    for (int v = 0; v < 8; ++v) {
        int k = frag_k(v, hb) + 32 * chunk;
        h2 t = *(const h2*)(p + k);
        a[2 * v] = t[0]; a[2 * v + 1] = t[1];
    }
    return a;
}

// ---------------- kernel 0: prep (weights->f16, qK fold, zero accumulators) ----------------
__global__ void prep_kernel(const float* ginW, const float* valW, const float* sencW,
                            const float* sdecW, const float* keyW, const float* keyb,
                            const float* qw, _Float16* w16, float* smallf) {
    int tid = threadIdx.x;
    for (int i = tid; i < 4096; i += 256) {
        w16[i]           = (_Float16)ginW[i];
        w16[4096 + i]    = (_Float16)valW[i];
        w16[2*4096 + i]  = (_Float16)sencW[i];
        w16[3*4096 + i]  = (_Float16)sdecW[i];
    }
    // qk[k] = sum_j q[j]*keyW[j][k]; qb = q . key_b
    if (tid < 64) {
        float acc = 0.f;
        for (int j = 0; j < 64; ++j) acc += qw[j] * keyW[j * 64 + tid];
        smallf[tid] = acc;                       // qk
    }
    if (tid == 0) {
        float acc = 0.f;
        for (int j = 0; j < 64; ++j) acc += qw[j] * keyb[j];
        smallf[64] = acc;                        // qb
    }
    // zero bnsum/bnsumsq/scale/shift/g_h/lossacc region: smallf[128 .. 898)
    for (int i = tid; i < 770; i += 256) smallf[128 + i] = 0.f;
}

// ---------------- kernel 1: encoder: h = gelu(x Wenc^T + b) + gene_emb ----------------
__global__ void encode_kernel(const float* __restrict__ x, const float* __restrict__ encW,
                              const float* __restrict__ encb, const float* __restrict__ gemb,
                              float* __restrict__ bufH, float* __restrict__ bufT) {
    int id = blockIdx.x * 256 + threadIdx.x;
    int node = id >> 2, part = id & 3;
    if (node >= NN) return;
    float xv[DIN];
    const float* xp = x + (size_t)node * DIN;
#pragma unroll
    for (int i = 0; i < DIN; ++i) xv[i] = xp[i];
    int r = node % NG;
    int fb = part * 16;
#pragma unroll
    for (int f = fb; f < fb + 16; ++f) {
        const float* wr = encW + f * DIN;
        float acc = encb[f];
#pragma unroll
        for (int i = 0; i < DIN; ++i) acc += xv[i] * wr[i];
        float h = gelu_exact(acc) + gemb[r * DH + f];
        bufH[(size_t)node * DH + f] = h;
        bufT[(size_t)node * DH + f] = h;
    }
}

// ---------------- kernel 2: edge scatter: bufT[dst] += bufH[src] ----------------
__global__ void edge_kernel(const int* __restrict__ src, const int* __restrict__ dst,
                            const float* __restrict__ bufH, float* __restrict__ bufT) {
    int id = blockIdx.x * 256 + threadIdx.x;
    int e = id >> 2, grp = id & 3;
    if (e >= EE) return;
    int s = src[e], d = dst[e];
    const float* sp = bufH + (size_t)s * DH + grp * 16;
    float* dp = bufT + (size_t)d * DH + grp * 16;
#pragma unroll
    for (int q = 0; q < 4; ++q) {
        float4 v = *(const float4*)(sp + 4 * q);
        atomicAdd(dp + 4 * q + 0, v.x);
        atomicAdd(dp + 4 * q + 1, v.y);
        atomicAdd(dp + 4 * q + 2, v.z);
        atomicAdd(dp + 4 * q + 3, v.w);
    }
}

// ---------------- kernel 3: GIN GEMM (WMMA) + BN statistics ----------------
__global__ void gin_kernel(const float* __restrict__ bufT, float* __restrict__ hpre,
                           const _Float16* __restrict__ w16gin,
                           float* __restrict__ bnsum, float* __restrict__ bnsumsq) {
    __shared__ _Float16 sW[4096];
    __shared__ float sSum[64], sSq[64];
    int tid = threadIdx.x;
    for (int i = tid; i < 4096; i += 256) sW[i] = w16gin[i];
    if (tid < 64) { sSum[tid] = 0.f; sSq[tid] = 0.f; }
    __syncthreads();

    int wave = tid >> 5, lane = tid & 31;
    int l = lane & 15, mh = (lane >> 4) * 8;
    int tile = blockIdx.x * 8 + wave;
    if (tile < NT) {
        const float* rowbase = bufT + (size_t)tile * 16 * DH;
        v16h a0 = load_a_gf32(rowbase, lane, 0);
        v16h a1 = load_a_gf32(rowbase, lane, 1);
#pragma unroll
        for (int ct = 0; ct < 4; ++ct) {
            v16h b0 = load_b_lds(sW, lane, ct * 16, 0);
            v16h b1 = load_b_lds(sW, lane, ct * 16, 1);
            v8f c = {};
            c = wmma16(a0, b0, c);
            c = wmma16(a1, b1, c);
            float s = 0.f, ss = 0.f;
            int col = ct * 16 + l;
#pragma unroll
            for (int v = 0; v < 8; ++v) {
                float val = c[v];
                hpre[(size_t)(tile * 16 + v + mh) * DH + col] = val;
                s += val; ss += val * val;
            }
            atomicAdd(&sSum[col], s);
            atomicAdd(&sSq[col], ss);
        }
    }
    __syncthreads();
    if (tid < 64) {
        atomicAdd(&bnsum[tid], sSum[tid]);
        atomicAdd(&bnsumsq[tid], sSq[tid]);
    }
}

// ---------------- kernel 4: BN finalize -> affine scale/shift ----------------
__global__ void bnfin_kernel(const float* bnsum, const float* bnsumsq,
                             const float* bnw, const float* bnb,
                             float* scale, float* shift) {
    int t = threadIdx.x;
    if (t < 64) {
        float invn = 1.0f / (float)NN;
        float mu = bnsum[t] * invn;
        float var = bnsumsq[t] * invn - mu * mu;
        float rs = rsqrtf(var + 1e-5f);
        float sc = bnw[t] * rs;
        scale[t] = sc;
        shift[t] = bnb[t] - mu * sc;
    }
}

// ---------------- kernel 5: fused tail per 16-row tile ----------------
__global__ void main_kernel(const float* __restrict__ hpre,
                            const float* __restrict__ scale, const float* __restrict__ shift,
                            const float* __restrict__ qk, const float* __restrict__ qbp,
                            const _Float16* __restrict__ w16val, const _Float16* __restrict__ w16enc,
                            const _Float16* __restrict__ w16dec,
                            const float* __restrict__ valb, const float* __restrict__ sencb,
                            const float* __restrict__ sdecb,
                            float* __restrict__ g_h, float* __restrict__ lossacc,
                            float* __restrict__ out) {
    __shared__ _Float16 sWv[4096], sWe[4096], sWd[4096];
    __shared__ _Float16 sZ[8][16 * DH];
    __shared__ float sSc[64], sSh[64], sQK[64];
    int tid = threadIdx.x;
    for (int i = tid; i < 4096; i += 256) {
        sWv[i] = w16val[i]; sWe[i] = w16enc[i]; sWd[i] = w16dec[i];
    }
    if (tid < 64) { sSc[tid] = scale[tid]; sSh[tid] = shift[tid]; sQK[tid] = qk[tid]; }
    __syncthreads();

    int wave = tid >> 5, lane = tid & 31;
    int l = lane & 15, mh = (lane >> 4) * 8;
    int tile = blockIdx.x * 8 + wave;
    bool valid = tile < NT;            // wave-uniform

    v16h a0, a1;
    float zre[4][8];
    float wm[8];
    float wl = 0.f;

    if (valid) {
        // h = gelu(hpre*scale + shift), packed directly into A-layout; fold qK dot
        const float* p = hpre + (size_t)tile * 16 * DH + l * DH;
        float dot = 0.f;
#pragma unroll
        for (int chunk = 0; chunk < 2; ++chunk) {
            v16h* ap = chunk ? &a1 : &a0;
#pragma unroll
            for (int v = 0; v < 8; ++v) {
                int k = frag_k(v, mh) + 32 * chunk;
                float2 t = *(const float2*)(p + k);
                float g0 = gelu_exact(t.x * sSc[k] + sSh[k]);
                float g1 = gelu_exact(t.y * sSc[k + 1] + sSh[k + 1]);
                dot += g0 * sQK[k] + g1 * sQK[k + 1];
                (*ap)[2 * v] = (_Float16)g0; (*ap)[2 * v + 1] = (_Float16)g1;
            }
        }
        dot += __shfl_xor(dot, 16, 32);              // full row dot (lanes l and l+16 agree)
        wl = 1.0f / (1.0f + expf(-(dot + qbp[0])));  // sigmoid
        if (mh == 0) out[OUT_W_OFF + tile * 16 + l] = wl;
#pragma unroll
        for (int v = 0; v < 8; ++v) wm[v] = __shfl(wl, v + mh, 32);  // w for row v+mh

        // v = h valW^T + valb ; z = v*w ; stage z (f16) for next A fragments
#pragma unroll
        for (int ct = 0; ct < 4; ++ct) {
            v16h b0 = load_b_lds(sWv, lane, ct * 16, 0);
            v16h b1 = load_b_lds(sWv, lane, ct * 16, 1);
            float bias = valb[ct * 16 + l];
            v8f c;
#pragma unroll
            for (int v = 0; v < 8; ++v) c[v] = bias;
            c = wmma16(a0, b0, c);
            c = wmma16(a1, b1, c);
#pragma unroll
            for (int v = 0; v < 8; ++v) {
                float z = c[v] * wm[v];
                zre[ct][v] = z;
                sZ[wave][(v + mh) * DH + ct * 16 + l] = (_Float16)z;
            }
        }
    }
    __syncthreads();

    float sse = 0.f, sabs = 0.f;
    if (valid) {
        v16h za0 = load_a_lds(sZ[wave], lane, 0);
        v16h za1 = load_a_lds(sZ[wave], lane, 1);
        // enc = relu(z sencW^T + b); overwrite stage buffer with enc
#pragma unroll
        for (int ct = 0; ct < 4; ++ct) {
            v16h b0 = load_b_lds(sWe, lane, ct * 16, 0);
            v16h b1 = load_b_lds(sWe, lane, ct * 16, 1);
            float bias = sencb[ct * 16 + l];
            v8f c;
#pragma unroll
            for (int v = 0; v < 8; ++v) c[v] = bias;
            c = wmma16(za0, b0, c);
            c = wmma16(za1, b1, c);
#pragma unroll
            for (int v = 0; v < 8; ++v) {
                float e = fmaxf(c[v], 0.f);
                sabs += e;
                sZ[wave][(v + mh) * DH + ct * 16 + l] = (_Float16)e;
            }
        }
        v16h ea0 = load_a_lds(sZ[wave], lane, 0);
        v16h ea1 = load_a_lds(sZ[wave], lane, 1);

        int g0 = (tile * 16) / NG;
        bool cross = ((tile * 16 + 15) / NG) != g0;
        // dec = enc sdecW^T + b; loss, store, graph pooling
#pragma unroll
        for (int ct = 0; ct < 4; ++ct) {
            v16h b0 = load_b_lds(sWd, lane, ct * 16, 0);
            v16h b1 = load_b_lds(sWd, lane, ct * 16, 1);
            float bias = sdecb[ct * 16 + l];
            v8f c;
#pragma unroll
            for (int v = 0; v < 8; ++v) c[v] = bias;
            c = wmma16(ea0, b0, c);
            c = wmma16(ea1, b1, c);
            int col = ct * 16 + l;
            float acc0 = 0.f, acc1 = 0.f;
#pragma unroll
            for (int v = 0; v < 8; ++v) {
                float d = c[v];
                float df = d - zre[ct][v];
                sse += df * df;
                int row = tile * 16 + v + mh;
                out[OUT_DEC_OFF + (size_t)row * DH + col] = d;
                float gc = d * wm[v];
                if (row / NG == g0) acc0 += gc; else acc1 += gc;
            }
            atomicAdd(&g_h[g0 * DH + col], acc0);
            if (cross) atomicAdd(&g_h[(g0 + 1) * DH + col], acc1);
        }
#pragma unroll
        for (int off = 16; off > 0; off >>= 1) {
            sse  += __shfl_xor(sse, off, 32);
            sabs += __shfl_xor(sabs, off, 32);
        }
        if (lane == 0) {
            atomicAdd(&lossacc[0], sse);
            atomicAdd(&lossacc[1], sabs);
        }
    }
}

// ---------------- kernel 6: predictor head + loss finalize ----------------
__global__ void final_kernel(const float* __restrict__ g_h, const float* __restrict__ lossacc,
                             const float* __restrict__ pW1, const float* __restrict__ pb1,
                             const float* __restrict__ pW2, const float* __restrict__ pb2,
                             float* __restrict__ out) {
    __shared__ float sp[8];
    int tid = threadIdx.x;
    if (tid < 8) sp[tid] = 0.f;
    __syncthreads();
    for (int idx = tid; idx < 512; idx += 256) {
        int g = idx >> 6, j = idx & 63;
        float acc = pb1[j];
#pragma unroll
        for (int k = 0; k < 64; ++k) acc += g_h[g * DH + k] * pW1[j * DH + k];
        atomicAdd(&sp[g], gelu_exact(acc) * pW2[j]);
    }
    __syncthreads();
    if (tid < 8) out[tid] = sp[tid] + pb2[0];
    if (tid == 0) out[OUT_LOSS_OFF] = (lossacc[0] + lossacc[1]) / (float)((size_t)NN * DH);
}

// ---------------- launch ----------------
extern "C" void kernel_launch(void* const* d_in, const int* in_sizes, int n_in,
                              void* d_out, int out_size, void* d_ws, size_t ws_size,
                              hipStream_t stream) {
    const float* x      = (const float*)d_in[0];
    const int*   src    = (const int*)d_in[1];
    const int*   dst    = (const int*)d_in[2];
    const float* encW   = (const float*)d_in[3];
    const float* encb   = (const float*)d_in[4];
    const float* gemb   = (const float*)d_in[5];
    const float* ginW   = (const float*)d_in[6];
    const float* bnw    = (const float*)d_in[7];
    const float* bnb    = (const float*)d_in[8];
    const float* keyW   = (const float*)d_in[9];
    const float* keyb   = (const float*)d_in[10];
    const float* qw     = (const float*)d_in[11];
    const float* valW   = (const float*)d_in[12];
    const float* valb   = (const float*)d_in[13];
    const float* sencW  = (const float*)d_in[14];
    const float* sencb  = (const float*)d_in[15];
    const float* sdecW  = (const float*)d_in[16];
    const float* sdecb  = (const float*)d_in[17];
    const float* pW1    = (const float*)d_in[18];
    const float* pb1    = (const float*)d_in[19];
    const float* pW2    = (const float*)d_in[20];
    const float* pb2    = (const float*)d_in[21];
    float* out = (float*)d_out;

    // workspace layout: bufA | bufT | w16[4*4096] | smallf[1024]
    float* bufA = (float*)d_ws;                         // NN*DH f32 (h, then hpre)
    float* bufT = bufA + (size_t)NN * DH;               // NN*DH f32 (h + agg)
    _Float16* w16 = (_Float16*)(bufT + (size_t)NN * DH);
    float* smallf   = (float*)(w16 + 4 * 4096);
    float* qk       = smallf;            // 64
    float* qb       = smallf + 64;       // 1
    float* bnsum    = smallf + 128;      // 64
    float* bnsumsq  = smallf + 192;      // 64
    float* scale    = smallf + 256;      // 64
    float* shift    = smallf + 320;      // 64
    float* g_h      = smallf + 384;      // 512
    float* lossacc  = smallf + 896;      // 2

    prep_kernel<<<1, 256, 0, stream>>>(ginW, valW, sencW, sdecW, keyW, keyb, qw, w16, smallf);
    encode_kernel<<<(NN * 4 + 255) / 256, 256, 0, stream>>>(x, encW, encb, gemb, bufA, bufT);
    edge_kernel<<<(EE * 4 + 255) / 256, 256, 0, stream>>>(src, dst, bufA, bufT);
    gin_kernel<<<(NT + 7) / 8, 256, 0, stream>>>(bufT, bufA, w16, bnsum, bnsumsq);
    bnfin_kernel<<<1, 64, 0, stream>>>(bnsum, bnsumsq, bnw, bnb, scale, shift);
    main_kernel<<<(NT + 7) / 8, 256, 0, stream>>>(bufA, scale, shift, qk, qb,
                                                  w16 + 4096, w16 + 2 * 4096, w16 + 3 * 4096,
                                                  valb, sencb, sdecb, g_h, lossacc, out);
    final_kernel<<<1, 256, 0, stream>>>(g_h, lossacc, pW1, pb1, pW2, pb2, out);
}
